// ScaledDotProductAttention_26912265076837
// MI455X (gfx1250) — compile-verified, matrix operand order checked by reading, not run on hardware
//
#include <hip/hip_runtime.h>
#include <hip/hip_bf16.h>

// ---------------------------------------------------------------------------
// MI455X (gfx1250) fused attention block:
//   qp = (q@Wq^T + bq)/8 ; kp = k@Wk^T + bk ; vp = v@Wv^T + bv      (bf16)
//   Sc = qp @ kp^T  (f32) ; P = softmax(Sc)  (bf16, in-place over Sc)
//   ctx = P @ vp    (bf16)
//   out = gelu(ctx@Wf^T + bf) + v            (f32)
// All GEMMs: v_wmma_f32_16x16x32_bf16 (wave32), fp32 accumulation.
// bf16 operand tiles staged with GLOBAL_LOAD_ASYNC_TO_LDS_B128 (ASYNCcnt),
// double-buffered LDS; fp32 operands staged via VGPR convert.
// Staging loops are fully unrolled (compile-time trip counts) -> branch-free.
// ---------------------------------------------------------------------------

typedef __attribute__((ext_vector_type(16))) __bf16 v16bf;
typedef __attribute__((ext_vector_type(8)))  __bf16 v8bf;
typedef __attribute__((ext_vector_type(8)))  float  v8f;

#define TEMPERATURE 8.0f

enum { EPI_F32 = 0, EPI_BF16 = 1, EPI_FINAL = 2 };

constexpr int LDSS = 48;   // LDS row stride in bf16: 96B rows -> 16B aligned

__device__ inline void wait_async0() {
  asm volatile("s_wait_asynccnt 0x0" ::: "memory");
}

// ---------------- tile staging: global -> LDS (64 rows x 32 cols) ----------
// 128 threads per block; all trip counts compile-time -> fully unrolled.

// fp32 input: load float4, convert, ds_store (cannot async-convert).
__device__ inline void stage_a(const float* __restrict__ g, long gStride,
                               __bf16* lds, int tid) {
  constexpr int CPR = 32 / 4;            // float4 chunks per row
#pragma unroll
  for (int i = 0; i < (64 * CPR) / 128; ++i) {   // 4 iterations
    const int c   = tid + i * 128;
    const int r   = c / CPR;
    const int col = (c % CPR) * 4;
    const float4 vv = *(const float4*)(g + (long)r * gStride + col);
    __bf16* d = lds + r * LDSS + col;
    d[0] = (__bf16)vv.x; d[1] = (__bf16)vv.y;
    d[2] = (__bf16)vv.z; d[3] = (__bf16)vv.w;
  }
}

// bf16 input: asynchronous cache->LDS DMA, 16B per lane, tracked by ASYNCcnt.
// GVS addressing: mem = SADDR(sgpr base) + VADDR(32-bit byte offset).
__device__ inline void stage_a(const __bf16* __restrict__ g, long gStride,
                               __bf16* lds, int tid) {
  constexpr int CPR = 32 / 8;            // 16B chunks per row
  const unsigned ldsBase = (unsigned)(uintptr_t)lds;
#pragma unroll
  for (int i = 0; i < (64 * CPR) / 128; ++i) {   // 2 iterations
    const int c    = tid + i * 128;
    const int r    = c / CPR;
    const int colB = (c % CPR) * 16;     // bytes within row
    unsigned voff  = (unsigned)((long)r * gStride * 2 + colB);
    unsigned laddr = ldsBase + (unsigned)(r * LDSS * 2 + colB);
    asm volatile("global_load_async_to_lds_b128 %0, %1, %2 offset:0"
                 :: "v"(laddr), "v"(voff), "s"(g) : "memory");
  }
}

// bf16 NN B-tile: global [32 k][64 n] -> LDS transposed [n][k] so the compute
// loop can use aligned ds_load_b128 fragments for every operand.
__device__ inline void stage_b_trans(const __bf16* __restrict__ g, long gStride,
                                     __bf16* lds, int tid) {
  constexpr int CPR = 64 / 8;            // 8-elem chunks per row
#pragma unroll
  for (int i = 0; i < (32 * CPR) / 128; ++i) {   // 2 iterations
    const int c   = tid + i * 128;
    const int r   = c / CPR;             // k
    const int col = (c % CPR) * 8;       // n base
    const v8bf vv = *(const v8bf*)(g + (long)r * gStride + col);
#pragma unroll
    for (int j = 0; j < 8; ++j)
      lds[(col + j) * LDSS + r] = vv[j];
  }
}

// ---------------- WMMA fragment load from LDS ------------------------------
// 16-bit A/B (16x32) per ISA: lanes 0-15 hold row 0..15, lanes 16-31 repeat
// rows with K offset +8; VGPR pairs 0-3 -> K 0..7, pairs 4-7 -> K 16..23.
// Row-major LDS [row][k]: two contiguous 16B reads per fragment.
__device__ inline v16bf frag_rowmajor(const __bf16* lds, int base, int lane) {
  int row = base + (lane & 15);
  int g   = lane >> 4;
  const __bf16* p = lds + row * LDSS + 8 * g;
  v8bf lo = *(const v8bf*)(p);        // K = 8g .. 8g+7
  v8bf hi = *(const v8bf*)(p + 16);   // K = 16+8g .. 23+8g
  v16bf f;
#pragma unroll
  for (int i = 0; i < 8; ++i) { f[i] = lo[i]; f[8 + i] = hi[i]; }
  return f;
}

// ---------------- generic batched WMMA GEMM --------------------------------
// C[M,N] = A[M,K] @ (TRANS_B ? B[N,K]^T : B[K,N])  + epilogue
// Block: 64x64 output tile, 128 threads (4 wave32), wave -> 32x32 quadrant.
// Double-buffered LDS: stage tile k+1 (async for bf16) while computing k.
template<typename AT, typename BT, int EPI, bool TRANS_B>
__global__ __launch_bounds__(128)
void gemm_wmma_kernel(const AT* __restrict__ A, long lda, long sA,
                      const BT* __restrict__ Bm, long ldb, long sB,
                      void* __restrict__ Cv, long ldc, long sC,
                      int K, const float* __restrict__ bias, float scale,
                      const float* __restrict__ residual)
{
  __shared__ __bf16 ldsA[2][64 * LDSS];
  __shared__ __bf16 ldsB[2][64 * LDSS];

  const int tid  = threadIdx.x;
  const int lane = tid & 31;
  const int wave = tid >> 5;
  const int wm   = (wave >> 1) * 32;
  const int wn   = (wave & 1) * 32;

  const long batch = blockIdx.z;
  const AT* Ab = A + batch * sA + (long)(blockIdx.y * 64) * lda;
  const BT* Bb = TRANS_B ? (Bm + batch * sB + (long)(blockIdx.x * 64) * ldb)
                         : (Bm + batch * sB + (long)(blockIdx.x * 64));

  auto stage = [&](int k0, int buf) {
    stage_a(Ab + k0, lda, ldsA[buf], tid);
    if constexpr (TRANS_B)
      stage_a(Bb + k0, ldb, ldsB[buf], tid);
    else
      stage_b_trans(Bb + (long)k0 * ldb, ldb, ldsB[buf], tid);
  };

  v8f acc[2][2] = {};

  stage(0, 0);
  wait_async0();
  __syncthreads();

  int cur = 0;
  for (int k0 = 0; k0 < K; k0 += 32) {
    if (k0 + 32 < K)
      stage(k0 + 32, cur ^ 1);   // overlap next-tile DMA with WMMA below

    v16bf a0 = frag_rowmajor(ldsA[cur], wm,      lane);
    v16bf a1 = frag_rowmajor(ldsA[cur], wm + 16, lane);
    v16bf b0 = frag_rowmajor(ldsB[cur], wn,      lane);
    v16bf b1 = frag_rowmajor(ldsB[cur], wn + 16, lane);

    acc[0][0] = __builtin_amdgcn_wmma_f32_16x16x32_bf16(false, a0, false, b0, (short)0, acc[0][0], false, false);
    acc[0][1] = __builtin_amdgcn_wmma_f32_16x16x32_bf16(false, a0, false, b1, (short)0, acc[0][1], false, false);
    acc[1][0] = __builtin_amdgcn_wmma_f32_16x16x32_bf16(false, a1, false, b0, (short)0, acc[1][0], false, false);
    acc[1][1] = __builtin_amdgcn_wmma_f32_16x16x32_bf16(false, a1, false, b1, (short)0, acc[1][1], false, false);

    wait_async0();      // my async stores into buf cur^1 have landed
    __syncthreads();    // everyone done reading cur / writing cur^1
    cur ^= 1;
  }

  // ---- epilogue (C/D layout: lane holds col n=lane&15; row r + 8*(lane>>4))
  const int nl = lane & 15, g = lane >> 4;
  const long mTile = (long)blockIdx.y * 64 + wm;
  const long nTile = (long)blockIdx.x * 64 + wn;
  const long cBase = batch * sC;
#pragma unroll
  for (int ti = 0; ti < 2; ++ti)
#pragma unroll
    for (int tj = 0; tj < 2; ++tj) {
      const long nc = nTile + 16 * tj + nl;
      float bv = 0.0f;
      if constexpr (EPI != EPI_F32) { if (bias) bv = bias[nc]; }
#pragma unroll
      for (int r = 0; r < 8; ++r) {
        const long mr  = mTile + 16 * ti + r + 8 * g;
        const long idx = cBase + mr * ldc + nc;
        const float x  = acc[ti][tj][r];
        if constexpr (EPI == EPI_F32) {
          ((float*)Cv)[idx] = x;
        } else if constexpr (EPI == EPI_BF16) {
          ((__bf16*)Cv)[idx] = (__bf16)((x + bv) * scale);
        } else {  // EPI_FINAL: exact GELU + residual
          const float y  = x + bv;
          const float ge = 0.5f * y * (1.0f + erff(y * 0.70710678118654752440f));
          ((float*)Cv)[idx] = ge + residual[mr * ldc + nc];
        }
      }
    }
}

// ---------------- row softmax: f32 scores -> bf16 probs (in place) --------
// One 256-thread block per row of 2048. Writes bf16 over the fp32 row; all
// reads complete before the reduction barriers, so aliasing is safe.
__global__ __launch_bounds__(256)
void softmax_kernel(const float* __restrict__ S, __bf16* __restrict__ P,
                    long srcStride, long dstStride)
{
  const long row = blockIdx.x;
  const float* s = S + row * srcStride;
  __bf16*      p = P + row * dstStride;
  __shared__ float red[256];

  float vals[8];
  float m = -3.4e38f;
#pragma unroll
  for (int i = 0; i < 8; ++i) {
    vals[i] = s[threadIdx.x + i * 256];
    m = fmaxf(m, vals[i]);
  }
  red[threadIdx.x] = m; __syncthreads();
  for (int off = 128; off > 0; off >>= 1) {
    if (threadIdx.x < off)
      red[threadIdx.x] = fmaxf(red[threadIdx.x], red[threadIdx.x + off]);
    __syncthreads();
  }
  m = red[0]; __syncthreads();

  float sum = 0.0f;
#pragma unroll
  for (int i = 0; i < 8; ++i) { vals[i] = __expf(vals[i] - m); sum += vals[i]; }
  red[threadIdx.x] = sum; __syncthreads();
  for (int off = 128; off > 0; off >>= 1) {
    if (threadIdx.x < off)
      red[threadIdx.x] += red[threadIdx.x + off];
    __syncthreads();
  }
  const float inv = 1.0f / red[0];
#pragma unroll
  for (int i = 0; i < 8; ++i)
    p[threadIdx.x + i * 256] = (__bf16)(vals[i] * inv);
}

// ---------------------------------------------------------------------------
extern "C" void kernel_launch(void* const* d_in, const int* in_sizes, int n_in,
                              void* d_out, int out_size, void* d_ws, size_t ws_size,
                              hipStream_t stream)
{
  const float* q   = (const float*)d_in[0];
  const float* k   = (const float*)d_in[1];
  const float* v   = (const float*)d_in[2];
  const float* Wq  = (const float*)d_in[3];
  const float* bq  = (const float*)d_in[4];
  const float* Wk  = (const float*)d_in[5];
  const float* bk  = (const float*)d_in[6];
  const float* Wv  = (const float*)d_in[7];
  const float* bv  = (const float*)d_in[8];
  const float* Wf  = (const float*)d_in[9];
  const float* bfp = (const float*)d_in[10];
  float* out = (float*)d_out;

  const int  B = 4, S = 2048, D = 1024;
  const long M = (long)B * S;  // 8192

  // Workspace layout (~134 MB):
  //   qp,kp,vp,ctx : bf16 [M,D] each            (4 * 16.78 MB)
  //   Sc           : f32  [B,S,S]               (67.1 MB) -- P bf16 aliases it
  __bf16* qp  = (__bf16*)d_ws;
  __bf16* kp  = qp  + M * D;
  __bf16* vp  = kp  + M * D;
  __bf16* ctx = vp  + M * D;
  float*  Sc  = (float*)(ctx + M * D);
  __bf16* P   = (__bf16*)Sc;            // row stride 2*S bf16 (in-place rows)

  dim3 blk(128);

  // 1) projections: [M,D] = X @ W^T + b   (qp pre-scaled by 1/TEMPERATURE)
  dim3 gProj(D / 64, M / 64, 1);
  gemm_wmma_kernel<float, float, EPI_BF16, true><<<gProj, blk, 0, stream>>>(
      q, D, 0, Wq, D, 0, (void*)qp, D, 0, D, bq, 1.0f / TEMPERATURE, nullptr);
  gemm_wmma_kernel<float, float, EPI_BF16, true><<<gProj, blk, 0, stream>>>(
      k, D, 0, Wk, D, 0, (void*)kp, D, 0, D, bk, 1.0f, nullptr);
  gemm_wmma_kernel<float, float, EPI_BF16, true><<<gProj, blk, 0, stream>>>(
      v, D, 0, Wv, D, 0, (void*)vp, D, 0, D, bv, 1.0f, nullptr);

  // 2) scores: Sc[b] = qp[b] @ kp[b]^T  (f32)
  dim3 gSc(S / 64, S / 64, B);
  gemm_wmma_kernel<__bf16, __bf16, EPI_F32, true><<<gSc, blk, 0, stream>>>(
      qp, D, (long)S * D, kp, D, (long)S * D,
      (void*)Sc, S, (long)S * S, D, nullptr, 1.0f, nullptr);

  // 3) softmax rows -> bf16 P (in place over Sc)
  softmax_kernel<<<dim3((unsigned)(B * S)), dim3(256), 0, stream>>>(
      Sc, P, (long)S, (long)2 * S);

  // 4) context: ctx[b] = P[b] @ vp[b]   (NN GEMM, bf16 out)
  dim3 gPV(D / 64, S / 64, B);
  gemm_wmma_kernel<__bf16, __bf16, EPI_BF16, false><<<gPV, blk, 0, stream>>>(
      P, (long)2 * S, (long)2 * S * S, vp, D, (long)S * D,
      (void*)ctx, D, (long)S * D, S, nullptr, 1.0f, nullptr);

  // 5) out = gelu(ctx @ Wf^T + bf) + v  (f32)
  dim3 gF(D / 64, M / 64, 1);
  gemm_wmma_kernel<__bf16, float, EPI_FINAL, true><<<gF, blk, 0, stream>>>(
      ctx, D, 0, Wf, D, 0, (void*)out, D, 0, D, bfp, 1.0f, v);
}
